// NeuroSymbolicClassifier_45509473468868
// MI455X (gfx1250) — compile-verified, matrix operand order checked by reading, not run on hardware
//
#include <hip/hip_runtime.h>
#include <math.h>

// Problem constants (match reference).
#define B_   8192
#define F_   64
#define C_   10
#define R_   4
#define H_   256
#define NH_  8
#define HD_  32
#define L_   2
#define MROWS (B_ * C_)          // 81920
#define EPSV 1e-5f

typedef __attribute__((ext_vector_type(16))) __bf16 v16bf;
typedef __attribute__((ext_vector_type(8)))  float  v8f;

union FragBF { v16bf v; uint4 q[2]; unsigned short u[16]; };

__device__ __forceinline__ float bf2f(unsigned short u) {
    unsigned int t = ((unsigned int)u) << 16;
    float f; __builtin_memcpy(&f, &t, 4); return f;
}
__device__ __forceinline__ unsigned short f2bf(float f) {
    unsigned int t; __builtin_memcpy(&t, &f, 4);
    t += 0x7FFFu + ((t >> 16) & 1u);          // round-to-nearest-even
    return (unsigned short)(t >> 16);
}
__device__ __forceinline__ float geluf(float x) {   // exact (erf) gelu
    return 0.5f * x * (1.0f + erff(x * 0.70710678118654752f));
}

__device__ float blk_sum(float v, float* s) {
    int t = threadIdx.x; s[t] = v; __syncthreads();
    for (int o = blockDim.x >> 1; o > 0; o >>= 1) {
        if (t < o) s[t] += s[t + o];
        __syncthreads();
    }
    float r = s[0]; __syncthreads(); return r;
}
__device__ float blk_max(float v, float* s) {
    int t = threadIdx.x; s[t] = v; __syncthreads();
    for (int o = blockDim.x >> 1; o > 0; o >>= 1) {
        if (t < o) s[t] = fmaxf(s[t], s[t + o]);
        __syncthreads();
    }
    float r = s[0]; __syncthreads(); return r;
}

// ---------- BatchNorm batch statistics: one block per feature ----------
__global__ void bn_stats_k(const float* __restrict__ x, float* mu, float* rs) {
    __shared__ float sh[256];
    int f = blockIdx.x;
    float s = 0.f, q = 0.f;
    for (int b = threadIdx.x; b < B_; b += 256) {
        float v = x[(size_t)b * F_ + f];
        s += v; q += v * v;
    }
    s = blk_sum(s, sh);
    q = blk_sum(q, sh);
    if (threadIdx.x == 0) {
        float m = s / (float)B_;
        float var = q / (float)B_ - m * m;   // biased batch variance
        mu[f] = m;
        rs[f] = rsqrtf(var + EPSV);
    }
}

// ---------- softmax over F for importance: one block (64 thr) per concept ----------
__global__ void softmax_row_k(const float* __restrict__ in, float* __restrict__ out) {
    __shared__ float sh[64];
    int c = blockIdx.x, t = threadIdx.x;
    float v = in[c * F_ + t];
    float mx = blk_max(v, sh);
    float e = expf(v - mx);
    float den = blk_sum(e, sh);
    out[c * F_ + t] = e / den;
}

// ---------- gate = softplus(steep) * tanh(direc) ----------
__global__ void gate_k(const float* __restrict__ steep, const float* __restrict__ direc,
                       float* __restrict__ gate, int n) {
    int i = blockIdx.x * blockDim.x + threadIdx.x;
    if (i < n) {
        float s = steep[i];
        float sp = (s > 20.f) ? s : log1pf(expf(s));
        gate[i] = sp * tanhf(direc[i]);
    }
}

// ---------- f32 -> bf16 weight conversion ----------
__global__ void cvt_bf16_k(const float* __restrict__ in, unsigned short* __restrict__ out, int n) {
    int i = blockIdx.x * blockDim.x + threadIdx.x;
    if (i < n) out[i] = f2bf(in[i]);
}

// ---------- rule bank: acts (B,C,F*R) in bf16 (A-operand of proj GEMM) ----------
__global__ void acts_k(const float* __restrict__ x, const float* __restrict__ mu,
                       const float* __restrict__ rs, const float* __restrict__ g,
                       const float* __restrict__ be, const float* __restrict__ w,
                       const float* __restrict__ gate, const float* __restrict__ thresh,
                       unsigned short* __restrict__ actsB) {
    int idx = blockIdx.x * 256 + threadIdx.x;    // b*C*F + c*F + f
    int f = idx % F_;
    int c = (idx / F_) % C_;
    int b = idx / (F_ * C_);
    float xn = (x[(size_t)b * F_ + f] - mu[f]) * rs[f] * g[f] + be[f];
    float xw = xn * w[c * F_ + f];
    int base = (c * F_ + f) * R_;
    unsigned short o[4];
#pragma unroll
    for (int r = 0; r < R_; ++r) {
        float z = gate[base + r] * (xw - thresh[base + r]);
        o[r] = f2bf(1.0f / (1.0f + expf(-z)));
    }
    uint2 pk;
    pk.x = (unsigned)o[0] | ((unsigned)o[1] << 16);
    pk.y = (unsigned)o[2] | ((unsigned)o[3] << 16);
    *(uint2*)(actsB + ((size_t)(b * C_ + c) * (F_ * R_) + f * R_)) = pk;
}

// ---------- rule_means: one block per (c, f*R+r) column ----------
__global__ void rule_mean_k(const unsigned short* __restrict__ actsB, float* __restrict__ out) {
    __shared__ float sh[256];
    int j = blockIdx.x;                 // c*256 + fr
    int c = j >> 8, fr = j & 255;
    float s = 0.f;
    for (int b = threadIdx.x; b < B_; b += 256)
        s += bf2f(actsB[((size_t)(b * C_ + c)) * 256 + fr]);
    s = blk_sum(s, sh);
    if (threadIdx.x == 0) out[j] = s / (float)B_;
}

// ---------- LayerNorm (row-wise) -> bf16 out: one block per row ----------
__global__ void ln_bf16_k(const float* __restrict__ x, const float* __restrict__ g,
                          const float* __restrict__ b, unsigned short* __restrict__ out, int N) {
    __shared__ float sh[256];
    size_t row = blockIdx.x;
    const float* xr = x + row * (size_t)N;
    float s = 0.f, q = 0.f;
    for (int i = threadIdx.x; i < N; i += 256) { float v = xr[i]; s += v; q += v * v; }
    s = blk_sum(s, sh);
    q = blk_sum(q, sh);
    float m = s / (float)N;
    float r = rsqrtf(q / (float)N - m * m + EPSV);
    unsigned short* orow = out + row * (size_t)N;
    for (int i = threadIdx.x; i < N; i += 256)
        orow[i] = f2bf((xr[i] - m) * r * g[i] + b[i]);
}

// ---------- WMMA bf16 GEMM: D[M,N] = A[M,K] * W[N,K]^T (+bias/gelu/resid) ----------
#define GF_GELU  1
#define GF_RESID 2
#define GF_OUTBF 4

__global__ void __launch_bounds__(256) gemm_bf16_k(
        const unsigned short* __restrict__ A, const unsigned short* __restrict__ W,
        const float* __restrict__ bias, const float* __restrict__ resid,
        float* __restrict__ outF, unsigned short* __restrict__ outB,
        int M, int N, int K, int flags) {
    int lane  = threadIdx.x & 31;
    int wave  = threadIdx.x >> 5;
    int m0    = blockIdx.x * 128 + wave * 16;      // 8 waves -> 128 rows per block
    int n0    = blockIdx.y * 64;                   // 4 N-tiles per wave
    int lhalf = lane >> 4;                         // K-half select (contiguous k = 16*lhalf + s, same for A and B)
    int l15   = lane & 15;

    const unsigned short* Ap = A + (size_t)(m0 + l15) * K + lhalf * 16;
    const unsigned short* Wp[4];
#pragma unroll
    for (int i = 0; i < 4; ++i)
        Wp[i] = W + (size_t)(n0 + i * 16 + l15) * K + lhalf * 16;

    v8f zero = {};
    v8f acc[4];
#pragma unroll
    for (int i = 0; i < 4; ++i) acc[i] = zero;

    for (int k = 0; k < K; k += 32) {
        FragBF a;
        a.q[0] = *(const uint4*)(Ap + k);
        a.q[1] = *(const uint4*)(Ap + k + 8);
        if (k + 32 < K) __builtin_prefetch(Ap + k + 32, 0, 1);  // global_prefetch next A tile
#pragma unroll
        for (int i = 0; i < 4; ++i) {
            FragBF bm;
            bm.q[0] = *(const uint4*)(Wp[i] + k);
            bm.q[1] = *(const uint4*)(Wp[i] + k + 8);
            acc[i] = __builtin_amdgcn_wmma_f32_16x16x32_bf16(
                false, a.v, false, bm.v, (short)0, acc[i], false, false);
        }
    }

    int mrow_off = lhalf * 8;                       // C/D layout: lanes 16..31 hold M+8
#pragma unroll
    for (int i = 0; i < 4; ++i) {
        int n = n0 + i * 16 + l15;
        float bv = bias ? bias[n] : 0.0f;
#pragma unroll
        for (int j = 0; j < 8; ++j) {
            int m = m0 + mrow_off + j;
            size_t off = (size_t)m * N + n;
            float v = acc[i][j] + bv;
            if (flags & GF_GELU)  v = geluf(v);
            if (flags & GF_RESID) v += resid[off];
            if (flags & GF_OUTBF) outB[off] = f2bf(v);
            else                  outF[off] = v;
        }
    }
}

// ---------- attention: one thread per (b, head, query) ----------
__global__ void attn_k(const float* __restrict__ qkv, unsigned short* __restrict__ outB) {
    int idx = blockIdx.x * 256 + threadIdx.x;   // b*NH*C + h*C + i
    int i = idx % C_;
    int h = (idx / C_) % NH_;
    int b = idx / (C_ * NH_);
    const float* base = qkv + (size_t)b * C_ * (3 * H_);
    float q[HD_];
    const float* qr = base + (size_t)i * (3 * H_) + h * HD_;
#pragma unroll
    for (int d = 0; d < HD_; ++d) q[d] = qr[d];
    float sc[C_];
    float mx = -3.4e38f;
    for (int j = 0; j < C_; ++j) {
        const float* kr = base + (size_t)j * (3 * H_) + H_ + h * HD_;
        float s = 0.f;
#pragma unroll
        for (int d = 0; d < HD_; ++d) s += q[d] * kr[d];
        s *= 0.17677669529663688f;              // 1/sqrt(32)
        sc[j] = s;
        mx = fmaxf(mx, s);
    }
    float den = 0.f;
    for (int j = 0; j < C_; ++j) { sc[j] = expf(sc[j] - mx); den += sc[j]; }
    float inv = 1.0f / den;
    unsigned short* orow = outB + ((size_t)(b * C_ + i)) * H_ + h * HD_;
    for (int d = 0; d < HD_; ++d) {
        float o = 0.f;
        for (int j = 0; j < C_; ++j)
            o += sc[j] * base[(size_t)j * (3 * H_) + 2 * H_ + h * HD_ + d];
        orow[d] = f2bf(o * inv);
    }
}

// ---------- classifier head: logits = hc @ W_c2^T + b_c2, / clip(temp) ----------
__global__ void head_k(const unsigned short* __restrict__ hc, const float* __restrict__ Wc2,
                       const float* __restrict__ bc2, const float* __restrict__ temp,
                       float* __restrict__ outLogits) {
    int idx = blockIdx.x * 256 + threadIdx.x;   // b*C + c
    int c = idx % C_;
    int b = idx / C_;
    float s = bc2[c];
    const unsigned short* hr = hc + (size_t)b * H_;
    const float* wr = Wc2 + c * H_;
    for (int k = 0; k < H_; ++k) s += bf2f(hr[k]) * wr[k];
    float t = fminf(fmaxf(temp[0], 0.5f), 2.0f);
    outLogits[idx] = s / t;
}

extern "C" void kernel_launch(void* const* d_in, const int* in_sizes, int n_in,
                              void* d_out, int out_size, void* d_ws, size_t ws_size,
                              hipStream_t stream) {
    (void)in_sizes; (void)n_in; (void)out_size; (void)ws_size;
    const float* x      = (const float*)d_in[0];
    const float* bng    = (const float*)d_in[1];
    const float* bnb    = (const float*)d_in[2];
    const float* imp    = (const float*)d_in[3];
    const float* thresh = (const float*)d_in[4];
    const float* steep  = (const float*)d_in[5];
    const float* direc  = (const float*)d_in[6];
    const float* W_proj = (const float*)d_in[7];
    const float* b_proj = (const float*)d_in[8];
    const float* ln1_g  = (const float*)d_in[9];
    const float* ln1_b  = (const float*)d_in[10];
    const float* W_qkv  = (const float*)d_in[11];
    const float* b_qkv  = (const float*)d_in[12];
    const float* W_o    = (const float*)d_in[13];
    const float* b_o    = (const float*)d_in[14];
    const float* ln2_g  = (const float*)d_in[15];
    const float* ln2_b  = (const float*)d_in[16];
    const float* W_ff1  = (const float*)d_in[17];
    const float* b_ff1  = (const float*)d_in[18];
    const float* W_ff2  = (const float*)d_in[19];
    const float* b_ff2  = (const float*)d_in[20];
    const float* lnc_g  = (const float*)d_in[21];
    const float* lnc_b  = (const float*)d_in[22];
    const float* W_c1   = (const float*)d_in[23];
    const float* b_c1   = (const float*)d_in[24];
    const float* W_c2   = (const float*)d_in[25];
    const float* b_c2   = (const float*)d_in[26];
    const float* temp   = (const float*)d_in[27];
    float* out = (float*)d_out;

    // ---- workspace carve-up ----
    char* ws = (char*)d_ws;
    size_t off = 0;
    auto take = [&](size_t bytes) -> char* {
        char* p = ws + off;
        off += bytes;
        off = (off + 255) & ~(size_t)255;
        return p;
    };
    float* mu   = (float*)take(F_ * 4);
    float* rs   = (float*)take(F_ * 4);
    float* wsm  = (float*)take(C_ * F_ * 4);
    float* gate = (float*)take(C_ * F_ * R_ * 4);
    unsigned short* WpB   = (unsigned short*)take((size_t)H_ * (F_ * R_) * 2);      // 65536
    unsigned short* WqkvB = (unsigned short*)take((size_t)L_ * 3 * H_ * H_ * 2);    // 393216
    unsigned short* WoB   = (unsigned short*)take((size_t)L_ * H_ * H_ * 2);        // 131072
    unsigned short* Wff1B = (unsigned short*)take((size_t)L_ * 4 * H_ * H_ * 2);    // 524288
    unsigned short* Wff2B = (unsigned short*)take((size_t)L_ * H_ * 4 * H_ * 2);    // 524288
    unsigned short* Wc1B  = (unsigned short*)take((size_t)H_ * (C_ * H_) * 2);      // 655360
    float*          feat  = (float*)take((size_t)MROWS * H_ * 4);                   // 80 MB
    unsigned short* Abf   = (unsigned short*)take((size_t)MROWS * H_ * 2);          // 40 MB (shared A-operand)
    float*          qkvF  = (float*)take((size_t)MROWS * 3 * H_ * 4);               // 240 MB (reused as bf16 FF-hidden)
    unsigned short* hcB   = (unsigned short*)take((size_t)B_ * H_ * 2);

    unsigned short* ffh = (unsigned short*)qkvF;   // FF hidden (bf16) overlays qkv after attention

    auto gemm = [&](const unsigned short* A, const unsigned short* W, const float* bias,
                    const float* resid, float* oF, unsigned short* oB,
                    int M, int N, int K, int flags) {
        dim3 g(M / 128, N / 64);
        gemm_bf16_k<<<g, dim3(256), 0, stream>>>(A, W, bias, resid, oF, oB, M, N, K, flags);
    };

    // ---- stage 0: stats / small precomputes / weight conversion ----
    bn_stats_k<<<F_, 256, 0, stream>>>(x, mu, rs);
    softmax_row_k<<<C_, 64, 0, stream>>>(imp, wsm);
    gate_k<<<(C_ * F_ * R_) / 256, 256, 0, stream>>>(steep, direc, gate, C_ * F_ * R_);
    cvt_bf16_k<<<(H_ * F_ * R_) / 256, 256, 0, stream>>>(W_proj, WpB, H_ * F_ * R_);
    cvt_bf16_k<<<(L_ * 3 * H_ * H_) / 256, 256, 0, stream>>>(W_qkv, WqkvB, L_ * 3 * H_ * H_);
    cvt_bf16_k<<<(L_ * H_ * H_) / 256, 256, 0, stream>>>(W_o, WoB, L_ * H_ * H_);
    cvt_bf16_k<<<(L_ * 4 * H_ * H_) / 256, 256, 0, stream>>>(W_ff1, Wff1B, L_ * 4 * H_ * H_);
    cvt_bf16_k<<<(L_ * H_ * 4 * H_) / 256, 256, 0, stream>>>(W_ff2, Wff2B, L_ * H_ * 4 * H_);
    cvt_bf16_k<<<(H_ * C_ * H_) / 256, 256, 0, stream>>>(W_c1, Wc1B, H_ * C_ * H_);

    // ---- rule bank + rule means ----
    acts_k<<<(B_ * C_ * F_) / 256, 256, 0, stream>>>(x, mu, rs, bng, bnb, wsm, gate, thresh, Abf);
    rule_mean_k<<<C_ * (F_ * R_), 256, 0, stream>>>(Abf, out + (size_t)B_ * C_);

    // ---- projection into transformer space ----
    gemm(Abf, WpB, b_proj, nullptr, feat, nullptr, MROWS, H_, F_ * R_, 0);

    // ---- two norm-first encoder layers ----
    for (int l = 0; l < L_; ++l) {
        ln_bf16_k<<<MROWS, 256, 0, stream>>>(feat, ln1_g + l * H_, ln1_b + l * H_, Abf, H_);
        gemm(Abf, WqkvB + (size_t)l * 3 * H_ * H_, b_qkv + l * 3 * H_, nullptr,
             qkvF, nullptr, MROWS, 3 * H_, H_, 0);
        attn_k<<<(B_ * NH_ * C_) / 256, 256, 0, stream>>>(qkvF, Abf);
        gemm(Abf, WoB + (size_t)l * H_ * H_, b_o + l * H_, feat,
             feat, nullptr, MROWS, H_, H_, GF_RESID);
        ln_bf16_k<<<MROWS, 256, 0, stream>>>(feat, ln2_g + l * H_, ln2_b + l * H_, Abf, H_);
        gemm(Abf, Wff1B + (size_t)l * 4 * H_ * H_, b_ff1 + l * 4 * H_, nullptr,
             nullptr, ffh, MROWS, 4 * H_, H_, GF_GELU | GF_OUTBF);
        gemm(ffh, Wff2B + (size_t)l * H_ * 4 * H_, b_ff2 + l * H_, feat,
             feat, nullptr, MROWS, H_, 4 * H_, GF_RESID);
    }

    // ---- classifier head ----
    ln_bf16_k<<<B_, 256, 0, stream>>>(feat, lnc_g, lnc_b, Abf, C_ * H_);   // (B, C*H) is contiguous view of feat
    gemm(Abf, Wc1B, b_c1, nullptr, nullptr, hcB, B_, H_, C_ * H_, GF_GELU | GF_OUTBF);
    head_k<<<(B_ * C_) / 256, 256, 0, stream>>>(hcB, W_c2, b_c2, temp, out);
}
